// LinearAttention_15324443312954
// MI455X (gfx1250) — compile-verified, matrix operand order checked by reading, not run on hardware
//
#include <hip/hip_runtime.h>
#include <hip/hip_bf16.h>

typedef __bf16 bf16;
typedef __attribute__((ext_vector_type(16))) __bf16 v16bf;
typedef __attribute__((ext_vector_type(8)))  __bf16 v8bf;
typedef __attribute__((ext_vector_type(8)))  float  v8f;
typedef __attribute__((ext_vector_type(4)))  int    v4i;

#define DIM    1024
#define HEADS  16
#define DHEAD  64
#define BATCH  4
#define SEQ    4096
#define ROWS   (BATCH*SEQ)   // 16384
#define BPITCH 40            // LDS row pitch (bf16): 80B = 20-bank stride, conflict-free b128

// ---------------------------------------------------------------------------
// CDNA5 async global -> LDS copy (ASYNCcnt-tracked). The builtin's first param
// is `int4 addrspace(1)*` (learned from the round-2 diagnostic); second is the
// LDS-side pointer. Guarded fallback: synchronous reg-staged copy.
// ---------------------------------------------------------------------------
#if __has_builtin(__builtin_amdgcn_global_load_async_to_lds_b128)
#define HAS_ASYNC_LDS 1
#else
#define HAS_ASYNC_LDS 0
#endif

typedef __attribute__((address_space(1))) v4i as1_v4i;
typedef __attribute__((address_space(3))) v4i as3_v4i;

__device__ __forceinline__ void async_cp_b128(const bf16* gsrc, bf16* ldst) {
#if HAS_ASYNC_LDS
    __builtin_amdgcn_global_load_async_to_lds_b128(
        (as1_v4i*)(void*)const_cast<bf16*>(gsrc),
        (as3_v4i*)(void*)ldst, 0, 0);
#else
    *(v8bf*)ldst = *(const v8bf*)gsrc;
#endif
}

__device__ __forceinline__ void wait_async0() {
#if HAS_ASYNC_LDS
#if __has_builtin(__builtin_amdgcn_s_wait_asynccnt)
    __builtin_amdgcn_s_wait_asynccnt(0);
#else
    asm volatile("s_wait_asynccnt 0x0" ::: "memory");
#endif
#endif
}

// ---------------------------------------------------------------------------
// bf16 WMMA fragment load: [16 rows][K] row-major panel, row baked into p.
// Lane L: hi = L/16 selects K sub-blocks {hi*8..} and {16+hi*8..}.
// ---------------------------------------------------------------------------
__device__ __forceinline__ v16bf frag_load(const bf16* __restrict__ p, int hi) {
    v8bf lo = *(const v8bf*)(p + hi * 8);
    v8bf hh = *(const v8bf*)(p + 16 + hi * 8);
    v16bf r;
#pragma unroll
    for (int i = 0; i < 8; ++i) { r[i] = lo[i]; r[i + 8] = hh[i]; }
    return r;
}

__device__ __forceinline__ v8f wmma_bf16(v16bf a, v16bf b, v8f c) {
    return __builtin_amdgcn_wmma_f32_16x16x32_bf16(
        false, a, false, b, (short)0, c, false, false);
}

// ---------------------------------------------------------------------------
// Kernel 1: fp32 -> bf16 elementwise convert
// ---------------------------------------------------------------------------
__global__ void k_cvt(const float* __restrict__ in, bf16* __restrict__ out, int n) {
    int i = blockIdx.x * blockDim.x + threadIdx.x;
    if (i < n) out[i] = (bf16)in[i];
}

// ---------------------------------------------------------------------------
// Kernel 2: W [K][N] fp32 row-major -> W^T [N][K] bf16 row-major (LDS tiled)
// ---------------------------------------------------------------------------
__global__ __launch_bounds__(256) void k_cvt_wT(const float* __restrict__ w,
                                                bf16* __restrict__ wT) {
    __shared__ float tile[32][33];
    int kb = blockIdx.x * 32, nb = blockIdx.y * 32;
    int tx = threadIdx.x & 31, ty = threadIdx.x >> 5;
#pragma unroll
    for (int r = ty; r < 32; r += 8) tile[r][tx] = w[(size_t)(kb + r) * DIM + nb + tx];
    __syncthreads();
#pragma unroll
    for (int r = ty; r < 32; r += 8) wT[(size_t)(nb + r) * DIM + kb + tx] = (bf16)tile[tx][r];
}

// ---------------------------------------------------------------------------
// Kernel 3: QKV projection + per-head softmax epilogue.
//  grid = (ROWS/128, HEADS, 3); block = 256 (8 waves); wave strip = 16x64.
//  B panel (64 x 32k, shared by all 8 waves) staged in LDS via async copies,
//  double-buffered on ASYNCcnt.
// ---------------------------------------------------------------------------
__global__ __launch_bounds__(256) void k_proj_qkv(
    const bf16* __restrict__ xb,
    const bf16* __restrict__ wqT, const bf16* __restrict__ wkT, const bf16* __restrict__ wvT,
    const float* __restrict__ bq, const float* __restrict__ bk, const float* __restrict__ bv,
    bf16* __restrict__ qout, bf16* __restrict__ kTbuf, bf16* __restrict__ vTbuf)
{
    const int mode = blockIdx.z;
    const bf16*  wT   = (mode == 0) ? wqT : (mode == 1) ? wkT : wvT;
    const float* bias = (mode == 0) ? bq  : (mode == 1) ? bk  : bv;

    const int lane = threadIdx.x & 31;
    const int wave = threadIdx.x >> 5;
    const int hi = lane >> 4, lo = lane & 15;
    const int rowBase = blockIdx.x * 128 + wave * 16;
    const int head    = blockIdx.y;
    const int colBase = head * DHEAD;

    // LDS staging for the shared B panel: 64 rows x 32 k, pitch 40 bf16.
    __shared__ bf16 sB[2][64 * BPITCH];
    const int tid = threadIdx.x;
    const int ldn = tid >> 2, ldp = tid & 3;           // 64 rows x 4 b128 chunks
    const bf16* gB = wT + (size_t)(colBase + ldn) * DIM + ldp * 8;
    bf16* lB0 = &sB[0][ldn * BPITCH + ldp * 8];
    bf16* lB1 = &sB[1][ldn * BPITCH + ldp * 8];

    v8f acc[4] = {};
    const bf16* arow = xb + (size_t)(rowBase + lo) * DIM;

    async_cp_b128(gB, lB0);
    int cur = 0;
    for (int s = 0; s < DIM / 32; ++s) {
        wait_async0();
        __syncthreads();                    // stage s resident; prev buffer free
        if (s + 1 < DIM / 32) async_cp_b128(gB + (size_t)(s + 1) * 32, cur ? lB0 : lB1);
        const int k0 = s * 32;
        __builtin_prefetch(arow + k0 + 128, 0, 0);
        v16bf a = frag_load(arow + k0, hi);
        const bf16* bbase = &sB[cur][0];
#pragma unroll
        for (int t = 0; t < 4; ++t) {
            v16bf b = frag_load(bbase + (t * 16 + lo) * BPITCH, hi);
            acc[t] = wmma_bf16(a, b, acc[t]);
        }
        cur ^= 1;
    }

    // bias (per output column n = colBase + t*16 + lo)
    float fac[4][8];
#pragma unroll
    for (int t = 0; t < 4; ++t) {
        float bv_ = bias[colBase + t * 16 + lo];
#pragma unroll
        for (int j = 0; j < 8; ++j) fac[t][j] = acc[t][j] + bv_;
    }

    if (mode < 2) {
        // softmax over 64 head-dims per row: 4 acc regs x 16-lane half reduce
#pragma unroll
        for (int j = 0; j < 8; ++j) {
            float m = fac[0][j];
#pragma unroll
            for (int t = 1; t < 4; ++t) m = fmaxf(m, fac[t][j]);
#pragma unroll
            for (int mk = 1; mk <= 8; mk <<= 1) m = fmaxf(m, __shfl_xor(m, mk, 32));
            float s = 0.f;
#pragma unroll
            for (int t = 0; t < 4; ++t) { fac[t][j] = __expf(fac[t][j] - m); s += fac[t][j]; }
#pragma unroll
            for (int mk = 1; mk <= 8; mk <<= 1) s += __shfl_xor(s, mk, 32);
            float inv = 1.0f / s;
#pragma unroll
            for (int t = 0; t < 4; ++t) fac[t][j] *= inv;
        }
    }

#pragma unroll
    for (int t = 0; t < 4; ++t) {
#pragma unroll
        for (int j = 0; j < 8; ++j) {
            int R = rowBase + j + 8 * hi;         // C layout: M = j + 8*(lane/16)
            int e = t * 16 + lo;                  //            N = lane%16
            bf16 val = (bf16)fac[t][j];
            if (mode == 0) {
                qout[(size_t)R * DIM + colBase + e] = val;
            } else {
                int b = R >> 12, tl = R & (SEQ - 1);
                bf16* dst = (mode == 1) ? kTbuf : vTbuf;
                dst[((size_t)(b * HEADS + head) * DHEAD + e) * SEQ + tl] = val;
            }
        }
    }
}

// ---------------------------------------------------------------------------
// Kernel 4: per (b,h): context^T = (k^T v)^T  (64x64, K=4096) + k_cumsum.
// ---------------------------------------------------------------------------
__global__ __launch_bounds__(256) void k_context(
    const bf16* __restrict__ kTbuf, const bf16* __restrict__ vTbuf,
    bf16* __restrict__ ctxT, float* __restrict__ ksum)
{
    const int p = blockIdx.x;                      // (b*16 + h)
    const int lane = threadIdx.x & 31, wave = threadIdx.x >> 5;
    const int hi = lane >> 4, lo = lane & 15;
    const bf16* kp = kTbuf + (size_t)p * DHEAD * SEQ;
    const bf16* vp = vTbuf + (size_t)p * DHEAD * SEQ;

    // ---- k_cumsum: 4 threads per row of 4096, LDS reduce ----
    {
        int row = threadIdx.x >> 2, q4 = threadIdx.x & 3;
        const bf16* r = kp + (size_t)row * SEQ + q4 * 1024;
        float s = 0.f;
        for (int t = 0; t < 1024; ++t) s += (float)r[t];
        __shared__ float red[256];
        red[threadIdx.x] = s;
        __syncthreads();
        if (q4 == 0)
            ksum[p * DHEAD + row] = red[threadIdx.x] + red[threadIdx.x + 1] +
                                    red[threadIdx.x + 2] + red[threadIdx.x + 3];
        __syncthreads();
    }

    const int dt = wave & 3;       // 16-wide d tile
    const int ep = wave >> 2;      // pair of 16-wide e tiles
    v8f acc[2] = {};
    const bf16* arow  = kp + (size_t)(dt * 16 + lo) * SEQ;
    const bf16* brow0 = vp + (size_t)(ep * 32 + lo) * SEQ;
    for (int k0 = 0; k0 < SEQ; k0 += 32) {
        __builtin_prefetch(arow + k0 + 256, 0, 0);
        v16bf a = frag_load(arow + k0, hi);
#pragma unroll
        for (int u = 0; u < 2; ++u) {
            v16bf b = frag_load(brow0 + (size_t)u * 16 * SEQ + k0, hi);
            acc[u] = wmma_bf16(a, b, acc[u]);
        }
    }
#pragma unroll
    for (int u = 0; u < 2; ++u)
#pragma unroll
        for (int j = 0; j < 8; ++j) {
            int d = dt * 16 + j + 8 * hi;
            int e = ep * 32 + u * 16 + lo;
            ctxT[(size_t)p * (DHEAD * DHEAD) + e * DHEAD + d] = (bf16)acc[u][j];
        }
}

// ---------------------------------------------------------------------------
// Kernel 5: y = (q @ ctx) * D_inv + q  per (b,h).  grid = (SEQ/128, 64).
// ---------------------------------------------------------------------------
__global__ __launch_bounds__(256) void k_apply(
    const bf16* __restrict__ qb, const bf16* __restrict__ ctxT,
    const float* __restrict__ ksum, bf16* __restrict__ yb)
{
    const int p = blockIdx.y, b = p >> 4, h = p & 15;
    const int lane = threadIdx.x & 31, wave = threadIdx.x >> 5;
    const int hi = lane >> 4, lo = lane & 15;
    const size_t rowBase = (size_t)b * SEQ + blockIdx.x * 128 + wave * 16;
    const bf16*  ctx = ctxT + (size_t)p * (DHEAD * DHEAD);
    const float* ks  = ksum + p * DHEAD;

    // D_inv for local row lo (both halves compute the same 16 values)
    const bf16* qr = qb + (rowBase + lo) * DIM + h * DHEAD;
    float s = 0.f;
#pragma unroll
    for (int d = 0; d < DHEAD; ++d) s += (float)qr[d] * ks[d];
    float dinv = 1.0f / s;

    v8f acc[4] = {};
    const bf16* arow  = qb  + (rowBase + lo) * DIM + h * DHEAD;
    const bf16* brow0 = ctx + lo * DHEAD;
#pragma unroll
    for (int k0 = 0; k0 < DHEAD; k0 += 32) {
        v16bf a = frag_load(arow + k0, hi);
#pragma unroll
        for (int t = 0; t < 4; ++t) {
            v16bf bb = frag_load(brow0 + (size_t)t * 16 * DHEAD + k0, hi);
            acc[t] = wmma_bf16(a, bb, acc[t]);
        }
    }

#pragma unroll
    for (int t = 0; t < 4; ++t)
#pragma unroll
        for (int j = 0; j < 8; ++j) {
            int M = j + 8 * hi;
            float di = __shfl(dinv, M, 32);       // lane M holds row M's D_inv
            size_t R = rowBase + M;
            int e = t * 16 + lo;
            float qv = (float)qb[R * DIM + h * DHEAD + e];
            yb[R * DIM + h * DHEAD + e] = (bf16)(acc[t][j] * di + qv);
        }
}

// ---------------------------------------------------------------------------
// Kernel 6: out = y @ Wp + bp (fp32 out).  grid = (ROWS/128, HEADS).
// Same async-staged B panel as the QKV kernel.
// ---------------------------------------------------------------------------
__global__ __launch_bounds__(256) void k_proj_out(
    const bf16* __restrict__ yb, const bf16* __restrict__ wpT,
    const float* __restrict__ bp, float* __restrict__ out)
{
    const int lane = threadIdx.x & 31, wave = threadIdx.x >> 5;
    const int hi = lane >> 4, lo = lane & 15;
    const int rowBase = blockIdx.x * 128 + wave * 16;
    const int colBase = blockIdx.y * 64;

    __shared__ bf16 sB[2][64 * BPITCH];
    const int tid = threadIdx.x;
    const int ldn = tid >> 2, ldp = tid & 3;
    const bf16* gB = wpT + (size_t)(colBase + ldn) * DIM + ldp * 8;
    bf16* lB0 = &sB[0][ldn * BPITCH + ldp * 8];
    bf16* lB1 = &sB[1][ldn * BPITCH + ldp * 8];

    v8f acc[4] = {};
    const bf16* arow = yb + (size_t)(rowBase + lo) * DIM;

    async_cp_b128(gB, lB0);
    int cur = 0;
    for (int s = 0; s < DIM / 32; ++s) {
        wait_async0();
        __syncthreads();
        if (s + 1 < DIM / 32) async_cp_b128(gB + (size_t)(s + 1) * 32, cur ? lB0 : lB1);
        const int k0 = s * 32;
        __builtin_prefetch(arow + k0 + 128, 0, 0);
        v16bf a = frag_load(arow + k0, hi);
        const bf16* bbase = &sB[cur][0];
#pragma unroll
        for (int t = 0; t < 4; ++t) {
            v16bf b = frag_load(bbase + (t * 16 + lo) * BPITCH, hi);
            acc[t] = wmma_bf16(a, b, acc[t]);
        }
        cur ^= 1;
    }

#pragma unroll
    for (int t = 0; t < 4; ++t) {
        float bv_ = bp[colBase + t * 16 + lo];
#pragma unroll
        for (int j = 0; j < 8; ++j) {
            int R = rowBase + j + 8 * hi;
            out[(size_t)R * DIM + colBase + t * 16 + lo] = acc[t][j] + bv_;
        }
    }
}

// ---------------------------------------------------------------------------
extern "C" void kernel_launch(void* const* d_in, const int* in_sizes, int n_in,
                              void* d_out, int out_size, void* d_ws, size_t ws_size,
                              hipStream_t stream) {
    const float* x  = (const float*)d_in[0];
    const float* Wq = (const float*)d_in[1];
    const float* bq = (const float*)d_in[2];
    const float* Wk = (const float*)d_in[3];
    const float* bk = (const float*)d_in[4];
    const float* Wv = (const float*)d_in[5];
    const float* bv = (const float*)d_in[6];
    const float* Wp = (const float*)d_in[7];
    const float* bp = (const float*)d_in[8];
    float* out = (float*)d_out;

    char* ws = (char*)d_ws;
    size_t off = 0;
    bf16* xb   = (bf16*)(ws + off); off += (size_t)ROWS * DIM * 2;       // reused as yb
    bf16* qb   = (bf16*)(ws + off); off += (size_t)ROWS * DIM * 2;
    bf16* kT   = (bf16*)(ws + off); off += (size_t)64 * DHEAD * SEQ * 2;
    bf16* vT   = (bf16*)(ws + off); off += (size_t)64 * DHEAD * SEQ * 2;
    bf16* wqT  = (bf16*)(ws + off); off += (size_t)DIM * DIM * 2;
    bf16* wkT  = (bf16*)(ws + off); off += (size_t)DIM * DIM * 2;
    bf16* wvT  = (bf16*)(ws + off); off += (size_t)DIM * DIM * 2;
    bf16* wpT  = (bf16*)(ws + off); off += (size_t)DIM * DIM * 2;
    bf16* ctxT = (bf16*)(ws + off); off += (size_t)64 * DHEAD * DHEAD * 2;
    float* ksum = (float*)(ws + off); off += (size_t)64 * DHEAD * 4;
    bf16* yb = xb;

    {
        int n = ROWS * DIM;
        k_cvt<<<dim3((n + 255) / 256), dim3(256), 0, stream>>>(x, xb, n);
    }
    {
        dim3 g(DIM / 32, DIM / 32), blk(256);
        k_cvt_wT<<<g, blk, 0, stream>>>(Wq, wqT);
        k_cvt_wT<<<g, blk, 0, stream>>>(Wk, wkT);
        k_cvt_wT<<<g, blk, 0, stream>>>(Wv, wvT);
        k_cvt_wT<<<g, blk, 0, stream>>>(Wp, wpT);
    }
    k_proj_qkv<<<dim3(ROWS / 128, HEADS, 3), dim3(256), 0, stream>>>(
        xb, wqT, wkT, wvT, bq, bk, bv, qb, kT, vT);
    k_context<<<dim3(BATCH * HEADS), dim3(256), 0, stream>>>(kT, vT, ctxT, ksum);
    k_apply<<<dim3(SEQ / 128, BATCH * HEADS), dim3(256), 0, stream>>>(qb, ctxT, ksum, yb);
    k_proj_out<<<dim3(ROWS / 128, HEADS), dim3(256), 0, stream>>>(yb, wpT, bp, out);
}
